// AttnLSTMDecoder_49735721287841
// MI455X (gfx1250) — matrix operand order repaired and checked by reading, not compiled, not run on hardware
//
#include <hip/hip_runtime.h>

#define L_SEQ 65536
#define STATE 100
#define ATT   100
#define EMB   100
#define VOCAB 128
#define TSTEPS 258

#define APAD  112     // ATT padded to 7*16
#define KPAD  224     // 2*STATE padded to 7*32

typedef __attribute__((ext_vector_type(16))) __bf16        v16bf;
typedef __attribute__((ext_vector_type(8)))  float         v8f;
typedef __attribute__((ext_vector_type(4)))  unsigned int  u32x4;

static __device__ __forceinline__ unsigned short f32_to_bf16(float f) {
    unsigned u = __float_as_uint(f);
    unsigned rounding = 0x7FFFu + ((u >> 16) & 1u);   // round-to-nearest-even
    return (unsigned short)((u + rounding) >> 16);
}

static __device__ __forceinline__ float bf16_to_f32(unsigned short s) {
    return __uint_as_float(((unsigned)s) << 16);
}

static __device__ __forceinline__ float sigmoidf(float x) {
    return 1.0f / (1.0f + __expf(-x));
}

// -------------------------------------------------------------------------
// Pad kernel: w1p[112][224] (bf16, zero-padded) from w1[100][200] (f32).
// One-time, 25088 elements.
// -------------------------------------------------------------------------
__global__ void __launch_bounds__(256)
pad_w1_kernel(const float* __restrict__ w1, unsigned short* __restrict__ w1p)
{
    const int idx = blockIdx.x * 256 + threadIdx.x;
    if (idx >= APAD * KPAD) return;
    const int a = idx / KPAD;
    const int k = idx % KPAD;
    unsigned short v = 0;
    if (a < ATT && k < 2 * STATE) v = f32_to_bf16(w1[a * (2 * STATE) + k]);
    w1p[idx] = v;
}

// -------------------------------------------------------------------------
// Kernel A: w1dtT[a][l] = sum_k w1[a][k] * input_mat[k][l]   (bf16 output)
//   GEMM  C[100 x 65536] = w1[100 x 200] x input_mat[200 x 65536]
//   One wave per 16-wide l tile, all 7 a-tiles, K in 7 chunks of 32.
//   A fragments: two unguarded b128 loads from padded bf16 w1p.
//   B fragments: unguarded f32 gathers for K-chunks 0..5; one divergent
//   branch only for the K tail (192..199).
// -------------------------------------------------------------------------
__global__ void __launch_bounds__(128)
w1dt_wmma_kernel(const float* __restrict__ input_mat,        // [200][L] f32
                 const unsigned short* __restrict__ w1p,     // [112][224] bf16
                 unsigned short* __restrict__ w1dtT)         // [100][L] bf16
{
    const int lane     = threadIdx.x & 31;
    const int waveId   = blockIdx.x * 4 + (threadIdx.x >> 5);  // 0..4095
    const int nbase    = waveId * 16;                          // l base
    const int laneHalf = lane >> 4;                            // 0 or 1
    const int laneMod  = lane & 15;

    v8f acc[7];
    #pragma unroll
    for (int m = 0; m < 7; ++m)
        #pragma unroll
        for (int i = 0; i < 8; ++i) acc[m][i] = 0.0f;

    // ---- full K chunks (0..5): no guards anywhere -----------------------
    for (int kc = 0; kc < 6; ++kc) {
        const int kbase = kc * 32;

        union { v16bf v; unsigned short s[16]; } B;
        #pragma unroll
        for (int j = 0; j < 16; ++j) {
            const int K = kbase + 8 * laneHalf + (j & 7) + 16 * (j >> 3);
            B.s[j] = f32_to_bf16(input_mat[(size_t)K * L_SEQ + nbase + laneMod]);
        }

        #pragma unroll
        for (int m = 0; m < 7; ++m) {
            const int arow = m * 16 + laneMod;
            // A layout: elements 0..7 -> K = kbase+8*laneHalf .. +7 (contiguous)
            //           elements 8..15 -> +16 further (contiguous)
            const unsigned short* ap = w1p + arow * KPAD + kbase + 8 * laneHalf;
            union { v16bf v; u32x4 q[2]; } A;
            A.q[0] = *(const u32x4*)(ap);
            A.q[1] = *(const u32x4*)(ap + 16);
            acc[m] = __builtin_amdgcn_wmma_f32_16x16x32_bf16(
                false, A.v, false, B.v, (short)0, acc[m], false, false);
        }
    }

    // ---- K tail chunk (kc == 6): valid K = 192..199 ---------------------
    {
        const int kbase = 192;
        union { v16bf v; unsigned short s[16]; } B;
        #pragma unroll
        for (int j = 0; j < 16; ++j) B.s[j] = 0;
        if (laneHalf == 0) {              // single divergent branch, 8 loads
            #pragma unroll
            for (int j = 0; j < 8; ++j)
                B.s[j] = f32_to_bf16(
                    input_mat[(size_t)(kbase + j) * L_SEQ + nbase + laneMod]);
        }

        #pragma unroll
        for (int m = 0; m < 7; ++m) {
            const int arow = m * 16 + laneMod;
            const unsigned short* ap = w1p + arow * KPAD + kbase + 8 * laneHalf;
            union { v16bf v; u32x4 q[2]; } A;     // padded zeros for K >= 200
            A.q[0] = *(const u32x4*)(ap);
            A.q[1] = *(const u32x4*)(ap + 16);
            acc[m] = __builtin_amdgcn_wmma_f32_16x16x32_bf16(
                false, A.v, false, B.v, (short)0, acc[m], false, false);
        }
    }

    // Store bf16: lane holds N = nbase+laneMod, VGPR v -> M = m*16+v+8*laneHalf
    #pragma unroll
    for (int m = 0; m < 7; ++m) {
        #pragma unroll
        for (int v = 0; v < 8; ++v) {
            const int M = m * 16 + v + 8 * laneHalf;
            if (M < ATT)
                w1dtT[(size_t)M * L_SEQ + nbase + laneMod] = f32_to_bf16(acc[m][v]);
        }
    }
}

// -------------------------------------------------------------------------
// Per-step kernel 1: e[l] = exp( v . tanh(w1dtT[:,l] + w2dt) + v_b )
// (softmax shift-invariant; |score| <= sum|v_w|+|v_b| ~ 8, so no max pass)
// 256 blocks x 256 threads == L exactly; coalesced bf16 w1dtT reads.
// -------------------------------------------------------------------------
__global__ void __launch_bounds__(256)
scores_kernel(const unsigned short* __restrict__ w1dtT,
              const float* __restrict__ w2dt,
              const float* __restrict__ v_w, const float* __restrict__ v_b,
              float* __restrict__ e, float* __restrict__ blockSum)
{
    __shared__ float sv[ATT], sw[ATT];
    __shared__ float red[256];
    const int tid = threadIdx.x;
    if (tid < ATT) { sv[tid] = v_w[tid]; sw[tid] = w2dt[tid]; }
    __syncthreads();

    const int l = blockIdx.x * 256 + tid;
    float s = v_b[0];
    #pragma unroll 4
    for (int a = 0; a < ATT; ++a)
        s += sv[a] * tanhf(bf16_to_f32(w1dtT[(size_t)a * L_SEQ + l]) + sw[a]);
    const float ev = __expf(s);
    e[l] = ev;

    red[tid] = ev;
    __syncthreads();
    for (int off = 128; off > 0; off >>= 1) {
        if (tid < off) red[tid] += red[tid + off];
        __syncthreads();
    }
    if (tid == 0) blockSum[blockIdx.x] = red[0];
}

__global__ void __launch_bounds__(256)
reduce_sum_kernel(const float* __restrict__ blockSum, float* __restrict__ gsum)
{
    __shared__ float red[256];
    const int tid = threadIdx.x;
    red[tid] = blockSum[tid];
    __syncthreads();
    for (int off = 128; off > 0; off >>= 1) {
        if (tid < off) red[tid] += red[tid + off];
        __syncthreads();
    }
    if (tid == 0) gsum[0] = red[0];
}

// -------------------------------------------------------------------------
// Per-step kernel 2: context[k] = (input_mat[k,:] . e) / gsum
// 100 blocks; each handles rows k and k+100, sharing e[] reads (float4).
// -------------------------------------------------------------------------
__global__ void __launch_bounds__(256)
context_kernel(const float* __restrict__ input_mat, const float* __restrict__ e,
               const float* __restrict__ gsum, float* __restrict__ context)
{
    const int k0 = blockIdx.x, k1 = blockIdx.x + STATE;
    const int tid = threadIdx.x;
    const float4* r0 = (const float4*)(input_mat + (size_t)k0 * L_SEQ);
    const float4* r1 = (const float4*)(input_mat + (size_t)k1 * L_SEQ);
    const float4* e4 = (const float4*)e;
    float s0 = 0.0f, s1 = 0.0f;
    for (int i = tid; i < L_SEQ / 4; i += 256) {
        const float4 ev = e4[i];
        const float4 a0 = r0[i];
        const float4 a1 = r1[i];
        s0 += a0.x * ev.x + a0.y * ev.y + a0.z * ev.z + a0.w * ev.w;
        s1 += a1.x * ev.x + a1.y * ev.y + a1.z * ev.z + a1.w * ev.w;
    }
    __shared__ float red0[256], red1[256];
    red0[tid] = s0; red1[tid] = s1;
    __syncthreads();
    for (int off = 128; off > 0; off >>= 1) {
        if (tid < off) { red0[tid] += red0[tid + off]; red1[tid] += red1[tid + off]; }
        __syncthreads();
    }
    if (tid == 0) {
        const float inv = 1.0f / gsum[0];
        context[k0] = red0[0] * inv;
        context[k1] = red1[0] * inv;
    }
}

// -------------------------------------------------------------------------
// Per-step kernel 3: LSTM cell + log-softmax loss + next-step w2dt.
// t == -1 : priming step (x = [0; emb[eos]], h=c=0, no loss).
// Single block (sequential dependency); 512 threads.
// -------------------------------------------------------------------------
__global__ void __launch_bounds__(512)
lstm_step_kernel(int t,
                 const int* __restrict__ output_ids, const int* __restrict__ eos_id,
                 const float* __restrict__ W_ih, const float* __restrict__ W_hh,
                 const float* __restrict__ b_ih, const float* __restrict__ b_hh,
                 const float* __restrict__ w2,
                 const float* __restrict__ lin_w, const float* __restrict__ lin_b,
                 const float* __restrict__ emb,
                 const float* __restrict__ context,
                 float* __restrict__ h, float* __restrict__ c,
                 float* __restrict__ w2dt,
                 float* __restrict__ loss_out)
{
    __shared__ float sx[3 * STATE];        // x = [context(200); last_emb(100)]
    __shared__ float sh[STATE], sc[STATE];
    __shared__ float gates[4 * STATE];
    __shared__ float hnew[STATE], cnew[STATE];
    __shared__ float slog[VOCAB];

    const int tid = threadIdx.x;
    const bool prime = (t < 0);
    const int prev_char = (t <= 0) ? eos_id[0] : output_ids[t - 1];

    if (tid < 2 * STATE)      sx[tid] = prime ? 0.0f : context[tid];
    else if (tid < 3 * STATE) sx[tid] = emb[prev_char * EMB + (tid - 2 * STATE)];
    if (tid < STATE) {
        sh[tid] = prime ? 0.0f : h[tid];
        sc[tid] = prime ? 0.0f : c[tid];
    }
    __syncthreads();

    if (tid < 4 * STATE) {
        float g = b_ih[tid] + b_hh[tid];
        const float* wi = W_ih + tid * (3 * STATE);
        #pragma unroll 4
        for (int k = 0; k < 3 * STATE; ++k) g += wi[k] * sx[k];
        const float* wh = W_hh + tid * STATE;
        #pragma unroll 4
        for (int k = 0; k < STATE; ++k) g += wh[k] * sh[k];
        gates[tid] = g;
    }
    __syncthreads();

    if (tid < STATE) {
        const float ig = sigmoidf(gates[tid]);
        const float fg = sigmoidf(gates[STATE + tid]);
        const float gg = tanhf(gates[2 * STATE + tid]);
        const float og = sigmoidf(gates[3 * STATE + tid]);
        const float cn = fg * sc[tid] + ig * gg;
        const float hn = og * tanhf(cn);
        cnew[tid] = cn; hnew[tid] = hn;
        c[tid] = cn;    h[tid] = hn;
    }
    __syncthreads();

    // w2dt for the next step's attention: w2 @ [h_new; c_new]
    if (tid < ATT) {
        const float* wr = w2 + tid * (2 * STATE);
        float s = 0.0f;
        #pragma unroll 4
        for (int k = 0; k < STATE; ++k) s += wr[k] * hnew[k];
        #pragma unroll 4
        for (int k = 0; k < STATE; ++k) s += wr[STATE + k] * cnew[k];
        w2dt[tid] = s;
    }

    if (!prime) {
        if (tid < VOCAB) {
            const float* lr = lin_w + tid * STATE;
            float s = lin_b[tid];
            #pragma unroll 4
            for (int k = 0; k < STATE; ++k) s += lr[k] * hnew[k];
            slog[tid] = s;
        }
        __syncthreads();
        if (tid == 0) {
            float m = slog[0];
            for (int j = 1; j < VOCAB; ++j) m = fmaxf(m, slog[j]);
            float sum = 0.0f;
            for (int j = 0; j < VOCAB; ++j) sum += __expf(slog[j] - m);
            const float lse = m + __logf(sum);
            const int ch = output_ids[t];
            const float loss_t = lse - slog[ch];
            loss_out[0] = (t == 0 ? 0.0f : loss_out[0]) + loss_t;  // fresh accum each call
        }
    }
}

// -------------------------------------------------------------------------
extern "C" void kernel_launch(void* const* d_in, const int* in_sizes, int n_in,
                              void* d_out, int out_size, void* d_ws, size_t ws_size,
                              hipStream_t stream)
{
    const float* input_mat  = (const float*)d_in[0];
    const int*   output_ids = (const int*)  d_in[1];
    const float* W_ih       = (const float*)d_in[2];
    const float* W_hh       = (const float*)d_in[3];
    const float* b_ih       = (const float*)d_in[4];
    const float* b_hh       = (const float*)d_in[5];
    const float* w1         = (const float*)d_in[6];
    const float* w2         = (const float*)d_in[7];
    const float* v_w        = (const float*)d_in[8];
    const float* v_b        = (const float*)d_in[9];
    const float* lin_w      = (const float*)d_in[10];
    const float* lin_b      = (const float*)d_in[11];
    const float* emb        = (const float*)d_in[12];
    const int*   eos_id     = (const int*)  d_in[13];

    // workspace layout (bytes)
    char* ws = (char*)d_ws;
    unsigned short* w1dtT = (unsigned short*)ws;                       // 100*65536 bf16
    unsigned short* w1p   = w1dtT + (size_t)ATT * L_SEQ;               // 112*224 bf16
    float* fbase    = (float*)(w1p + APAD * KPAD);
    float* e        = fbase;                                           // 65536
    float* blockSum = e + L_SEQ;                                       // 256
    float* gsum     = blockSum + 256;                                  // 1
    float* context  = gsum + 1;                                        // 200
    float* h        = context + 2 * STATE;                             // 100
    float* c        = h + STATE;                                       // 100
    float* w2dt     = c + STATE;                                       // 100

    float* loss = (float*)d_out;

    // One-time: pad w1 -> bf16, then WMMA GEMM (w1dtT stays L2-resident)
    pad_w1_kernel<<<(APAD * KPAD + 255) / 256, 256, 0, stream>>>(w1, w1p);
    w1dt_wmma_kernel<<<1024, 128, 0, stream>>>(input_mat, w1p, w1dtT);

    // Priming LSTM step (t = -1) also produces w2dt for step 0
    lstm_step_kernel<<<1, 512, 0, stream>>>(-1, output_ids, eos_id,
                                            W_ih, W_hh, b_ih, b_hh, w2,
                                            lin_w, lin_b, emb,
                                            context, h, c, w2dt, loss);

    for (int t = 0; t < TSTEPS; ++t) {
        scores_kernel<<<256, 256, 0, stream>>>(w1dtT, w2dt, v_w, v_b, e, blockSum);
        reduce_sum_kernel<<<1, 256, 0, stream>>>(blockSum, gsum);
        context_kernel<<<STATE, 256, 0, stream>>>(input_mat, e, gsum, context);
        lstm_step_kernel<<<1, 512, 0, stream>>>(t, output_ids, eos_id,
                                                W_ih, W_hh, b_ih, b_hh, w2,
                                                lin_w, lin_b, emb,
                                                context, h, c, w2dt, loss);
    }
}
